// SupernodePooling_50130858278962
// MI455X (gfx1250) — matrix operand order repaired and checked by reading, not compiled, not run on hardware
//
#include <hip/hip_runtime.h>
#include <hip/hip_bf16.h>
#include <math.h>

// Problem constants from the reference
#define BB   2
#define NN   8192
#define SS   1024
#define KK   32
#define HID  256
#define MTOT (BB * SS * KK)   // 65536 rows through the MLP

typedef __attribute__((ext_vector_type(16))) _Float16 v16h;
typedef __attribute__((ext_vector_type(8)))  _Float16 v8h;
typedef __attribute__((ext_vector_type(8)))  float    v8f;

// ---------------------------------------------------------------- KNN
// One block per (b, s). Distances in LDS, then 32 stable argmin passes
// (tie -> smaller index, matching jnp.argsort stability).
__global__ __launch_bounds__(256) void sp_knn_kernel(
    const float* __restrict__ pos, const int* __restrict__ sup_idx,
    float* __restrict__ coords) {
    __shared__ float dist[NN];
    __shared__ float sval[256];
    __shared__ int   sidx[256];

    const int t  = threadIdx.x;
    const int bs = blockIdx.x;          // 0 .. B*S-1
    const int b  = bs >> 10;            // / S
    const int s  = bs & (SS - 1);

    const float* X = pos + (size_t)b * NN * 3;
    const int sup = sup_idx[s];
    const float sx = X[sup * 3 + 0];
    const float sy = X[sup * 3 + 1];
    const float sz = X[sup * 3 + 2];

    for (int p = t; p < NN; p += 256) {
        float dx = X[p * 3 + 0] - sx;
        float dy = X[p * 3 + 1] - sy;
        float dz = X[p * 3 + 2] - sz;
        dist[p] = dx * dx + dy * dy + dz * dz;  // sqrt is monotonic: skip it
    }
    __syncthreads();

    for (int k = 0; k < KK; ++k) {
        float bv = INFINITY; int bi = NN;
        for (int p = t; p < NN; p += 256) {
            float v = dist[p];
            if (v < bv || (v == bv && p < bi)) { bv = v; bi = p; }
        }
        sval[t] = bv; sidx[t] = bi;
        __syncthreads();
        for (int off = 128; off > 0; off >>= 1) {
            if (t < off) {
                float ov = sval[t + off]; int oi = sidx[t + off];
                if (ov < sval[t] || (ov == sval[t] && oi < sidx[t])) {
                    sval[t] = ov; sidx[t] = oi;
                }
            }
            __syncthreads();
        }
        if (t == 0) {
            int w = sidx[0];
            float* oc = coords + ((size_t)bs * KK + k) * 3;
            oc[0] = X[w * 3 + 0];
            oc[1] = X[w * 3 + 1];
            oc[2] = X[w * 3 + 2];
            dist[w] = INFINITY;
        }
        __syncthreads();
    }
}

// ---------------------------------------------------------------- weight transpose + f16 cast
// wT[n][k] = (f16) w[k][n], 256x256
__global__ void sp_transpose_kernel(const float* __restrict__ w, _Float16* __restrict__ wT) {
    int idx = blockIdx.x * 256 + threadIdx.x;   // 65536 total
    int n = idx & 255;
    int k = idx >> 8;
    wT[n * 256 + k] = (_Float16)w[k * 256 + n];
}

// ---------------------------------------------------------------- features: proj + sincos PE
// One block per row r (65536 rows), thread = output channel c.
__global__ __launch_bounds__(256) void sp_feature_kernel(
    const float* __restrict__ coords, const float* __restrict__ w_in,
    const float* __restrict__ b_in, _Float16* __restrict__ h0) {
    const int r = blockIdx.x;
    const int c = threadIdx.x;
    const float c0 = coords[(size_t)r * 3 + 0];
    const float c1 = coords[(size_t)r * 3 + 1];
    const float c2 = coords[(size_t)r * 3 + 2];

    float v = b_in[c] + c0 * w_in[c] + c1 * w_in[256 + c] + c2 * w_in[512 + c];

    // HID=256, ndim=3 -> 85 per dim, half=42 -> 3*84=252 used, 4 zero-pad
    if (c < 252) {
        int i = c / 84;
        int j = c % 84;
        int jj = (j < 42) ? j : j - 42;
        // freq[jj] = exp(-jj * ln(10000)/41); args are O(1): fast trans ops OK
        float f = __expf((float)jj * (-9.210340371976184f / 41.0f));
        float cd = (i == 0) ? c0 : ((i == 1) ? c1 : c2);
        float e = cd * f;
        v += (j < 42) ? __sinf(e) : __cosf(e);
    }
    h0[(size_t)r * HID + c] = (_Float16)v;
}

// ---------------------------------------------------------------- WMMA GEMM
// 256 threads = 8 waves; each wave owns 32 rows (one supernode) x 256 cols:
// two 16x16 C tiles share every B fragment (2 WMMAs per ds_load pair).
// Weight tiles (16 cols x 256 K = 8KB) are double-buffered in LDS and
// prefetched with GLOBAL_LOAD_ASYNC_TO_LDS_B128, fenced by s_wait_asynccnt.
// MODE is a template arg so the fully-unrolled nt loop is branch-free:
//   MODE 0: Aout[r] = gelu(row)      (GEMM-1, in-place over A)
//   MODE 1: outMean[sn] = mean_K     (GEMM-2, direct store, no atomics)
static __device__ inline v16h sp_cat8(v8h lo, v8h hi) {
    return __builtin_shufflevector(lo, hi, 0, 1, 2, 3, 4, 5, 6, 7,
                                   8, 9, 10, 11, 12, 13, 14, 15);
}

static __device__ inline float sp_gelu(float x) {
    // tanh-gelu == x * sigmoid(2 * 0.79788456 * (x + 0.044715 x^3))
    float t = 1.5957691216057308f * (x + 0.044715f * x * x * x);
    return __fdividef(x, 1.0f + __expf(-t));
}

template <int MODE>
__global__ __launch_bounds__(256) void sp_gemm_kernel(
    const _Float16* A, const _Float16* __restrict__ WT,
    const float* __restrict__ bias, _Float16* Aout, float* outMean) {
    __shared__ _Float16 sB[2][16][264];   // 2 x (16 cols x 256 K, +8 pad)

    const int lane = threadIdx.x & 31;
    const int lrow = lane & 15;
    const bool hiHalf = lane >= 16;
    const int m0 = blockIdx.x * 256 + (threadIdx.x >> 5) * 32;  // rows m0..m0+31

    const int stRow = threadIdx.x >> 4;          // staging: 16 threads per row
    const int stK   = (threadIdx.x & 15) * 16;   // 16 halves (32B) per thread
    const unsigned long long wtBase = (unsigned long long)(size_t)WT;

    // async-stage weight tile nt into LDS buffer buf (each thread moves 32B)
    auto stage = [&](int nt_, int buf_) {
        unsigned lds0 = (unsigned)(size_t)&sB[buf_][stRow][stK];
        unsigned voff = (unsigned)(((nt_ * 16 + stRow) * HID + stK) * 2);
        asm volatile(
            "global_load_async_to_lds_b128 %0, %1, %2\n\t"
            "global_load_async_to_lds_b128 %0, %1, %2 offset:16"
            :: "v"(lds0), "v"(voff), "s"(wtBase)
            : "memory");
    };

    stage(0, 0);   // kick off tile 0 before loading A (overlap)

    // Load all sixteen A fragments for this wave's 32 rows (registers).
    const _Float16* ar0 = A + (size_t)(m0 + lrow) * HID;
    const _Float16* ar1 = A + (size_t)(m0 + 16 + lrow) * HID;
    v16h a0[8], a1[8];
#pragma unroll
    for (int kt = 0; kt < 8; ++kt) {
        int kbase = kt * 32 + (hiHalf ? 8 : 0);
        a0[kt] = sp_cat8(*(const v8h*)(ar0 + kbase), *(const v8h*)(ar0 + kbase + 16));
        a1[kt] = sp_cat8(*(const v8h*)(ar1 + kbase), *(const v8h*)(ar1 + kbase + 16));
    }

    // Preload per-lane bias for all 16 n-tiles (kills per-nt load bubble).
    float bv[16];
#pragma unroll
    for (int nt = 0; nt < 16; ++nt) bv[nt] = bias[nt * 16 + lrow];

    asm volatile("s_wait_asynccnt 0x0" ::: "memory");
    __syncthreads();

#pragma unroll
    for (int nt = 0; nt < 16; ++nt) {           // fully unrolled: cur, LDS
        const int cur = nt & 1;                 // addresses, bv[nt] all static
        if (nt < 15) stage(nt + 1, cur ^ 1);    // prefetch next tile

        float bn = bv[nt];
        v8f acc0 = {bn, bn, bn, bn, bn, bn, bn, bn};
        v8f acc1 = acc0;

#pragma unroll
        for (int kt = 0; kt < 8; ++kt) {
            int kb = kt * 32 + (hiHalf ? 8 : 0);
            v16h bfrag = sp_cat8(*(const v8h*)(&sB[cur][lrow][kb]),
                                 *(const v8h*)(&sB[cur][lrow][kb + 16]));
            acc0 = __builtin_amdgcn_wmma_f32_16x16x32_f16(
                false, a0[kt], false, bfrag, (short)0, acc0, false, false);
            acc1 = __builtin_amdgcn_wmma_f32_16x16x32_f16(
                false, a1[kt], false, bfrag, (short)0, acc1, false, false);
        }

        if (MODE == 0) {
#pragma unroll
            for (int i = 0; i < 8; ++i) {
                int r0 = m0 + i + (hiHalf ? 8 : 0);
                Aout[(size_t)r0 * HID + nt * 16 + lrow]        = (_Float16)sp_gelu(acc0[i]);
                Aout[(size_t)(r0 + 16) * HID + nt * 16 + lrow] = (_Float16)sp_gelu(acc1[i]);
            }
        } else {
            float p = 0.0f;
#pragma unroll
            for (int i = 0; i < 8; ++i) p += acc0[i] + acc1[i];  // 16 of 32 rows
            p += __shfl_xor(p, 16, 32);                          // all 32 rows
            if (!hiHalf)  // wave owns the whole supernode: direct store
                outMean[(size_t)(m0 >> 5) * HID + nt * 16 + lrow] = p * (1.0f / 32.0f);
        }

        if (nt < 15) {
            asm volatile("s_wait_asynccnt 0x0" ::: "memory");
            __syncthreads();
        }
    }
}

// ---------------------------------------------------------------- launch
extern "C" void kernel_launch(void* const* d_in, const int* in_sizes, int n_in,
                              void* d_out, int out_size, void* d_ws, size_t ws_size,
                              hipStream_t stream) {
    (void)in_sizes; (void)n_in; (void)out_size; (void)ws_size;
    const float* input_pos = (const float*)d_in[0];   // (B,N,3)
    const int*   sup_idx   = (const int*)d_in[1];     // (S,)
    const float* w_in      = (const float*)d_in[2];   // (3,256)
    const float* b_in      = (const float*)d_in[3];   // (256,)
    const float* w1        = (const float*)d_in[4];   // (256,256)
    const float* b1        = (const float*)d_in[5];
    const float* w2        = (const float*)d_in[6];
    const float* b2        = (const float*)d_in[7];
    float* out = (float*)d_out;                       // (B,S,256) fp32

    char* ws = (char*)d_ws;
    float*    coords = (float*)ws;                               // 786432 B
    _Float16* h0     = (_Float16*)(ws + 786432);                 // 33554432 B (reused for h1)
    _Float16* w1T    = (_Float16*)(ws + 786432 + 33554432);      // 131072 B
    _Float16* w2T    = w1T + HID * HID;                          // 131072 B

    sp_knn_kernel<<<BB * SS, 256, 0, stream>>>(input_pos, sup_idx, coords);
    sp_transpose_kernel<<<256, 256, 0, stream>>>(w1, w1T);
    sp_transpose_kernel<<<256, 256, 0, stream>>>(w2, w2T);
    sp_feature_kernel<<<MTOT, 256, 0, stream>>>(coords, w_in, b_in, h0);
    // h1 = gelu(h0 @ w1 + b1), written in place over h0
    sp_gemm_kernel<0><<<MTOT / 256, 256, 0, stream>>>(h0, w1T, b1, h0, nullptr);
    // out = mean_K(h1 @ w2 + b2), direct store (wave owns a supernode)
    sp_gemm_kernel<1><<<MTOT / 256, 256, 0, stream>>>(h0, w2T, b2, nullptr, out);
}
